// ContrastiveLoss_75909251990177
// MI455X (gfx1250) — matrix operand order repaired and checked by reading, not compiled, not run on hardware
//
#include <hip/hip_runtime.h>
#include <hip/hip_bf16.h>

typedef __attribute__((ext_vector_type(16))) _Float16 v16h;
typedef __attribute__((ext_vector_type(8)))  _Float16 v8h;
typedef __attribute__((ext_vector_type(4)))  _Float16 v4h;
typedef __attribute__((ext_vector_type(8)))  float    v8f;

#define B_ROWS   4096
#define NROWS    8192          // 2B
#define DIM      512
#define INV_T    2.0f          // 1 / TEMPERATURE
#define N_STRIPS (NROWS / 64)  // 128 strips of 4 N-tiles
#define M_BLOCK  32            // rows per block (2 M tiles)
#define WAVES    8

union FragU { v16h v; v8h h[2]; };

// ---------------------------------------------------------------------------
// Kernel 1: row-normalize z = concat(f1, f2); emit f16 zn[8192][512].
// One wave32 per row; each lane owns 16 floats (4x float4).
// ---------------------------------------------------------------------------
__global__ void nt_normalize(const float* __restrict__ f1,
                             const float* __restrict__ f2,
                             _Float16* __restrict__ zn) {
  const int wave = threadIdx.x >> 5;
  const int lane = threadIdx.x & 31;
  const int row  = blockIdx.x * WAVES + wave;

  const float* src = (row < B_ROWS) ? (f1 + (size_t)row * DIM)
                                    : (f2 + (size_t)(row - B_ROWS) * DIM);
  float4 vals[4];
  float ss = 0.f;
#pragma unroll
  for (int i = 0; i < 4; ++i) {
    const int col = i * 128 + lane * 4;
    vals[i] = *(const float4*)(src + col);
    ss += vals[i].x * vals[i].x + vals[i].y * vals[i].y +
          vals[i].z * vals[i].z + vals[i].w * vals[i].w;
  }
#pragma unroll
  for (int off = 16; off >= 1; off >>= 1) ss += __shfl_xor(ss, off, 32);

  const float inv = 1.0f / fmaxf(sqrtf(ss), 1e-8f);  // torch-style norm clamp
  _Float16* dst = zn + (size_t)row * DIM;
#pragma unroll
  for (int i = 0; i < 4; ++i) {
    const int col = i * 128 + lane * 4;
    v4h h = { (_Float16)(vals[i].x * inv), (_Float16)(vals[i].y * inv),
              (_Float16)(vals[i].z * inv), (_Float16)(vals[i].w * inv) };
    *(v4h*)(dst + col) = h;
  }
}

// A fragment: 16x32 f16, lane(0-15)=M; half 0 -> K{0-7,16-23}, half 1 -> K{8-15,24-31}
__device__ __forceinline__ v16h load_a(const _Float16* arow, int k0, int hi) {
  FragU a;
  a.h[0] = *(const v8h*)(arow + k0 + hi * 8);
  a.h[1] = *(const v8h*)(arow + k0 + 16 + hi * 8);
  return a.v;
}
// B fragment: 32x16 f16, lane(0-15)=N; half 0 -> K{0-7,16-23}, half 1 -> K{8-15,24-31}
__device__ __forceinline__ v16h load_b(const _Float16* brow, int k0, int hi) {
  FragU b;
  b.h[0] = *(const v8h*)(brow + k0 + hi * 8);
  b.h[1] = *(const v8h*)(brow + k0 + 16 + hi * 8);
  return b.v;
}

#define WMMA(a, b, c) \
  __builtin_amdgcn_wmma_f32_16x16x32_f16(false, (a), false, (b), (short)0, (c), false, false)

// ---------------------------------------------------------------------------
// Kernel 2: fused zn @ zn^T (WMMA f16 -> f32) with online sum-exp per row.
// Block = 8 waves owning 32 rows (2 M tiles). Each wave computes 32x64 output
// strips (2 M x 4 N tiles): per k-step 12 b128 loads feed 8 WMMAs (1.5/WMMA).
// logits bounded by |cos|/T <= 2, so single-pass sum-exp is stable (no max).
// Positive logits = diagonals of two known tiles; waves 0/1 recompute them
// after the branch-free hot loop.
// ---------------------------------------------------------------------------
__global__ void nt_gemm_softmax(const _Float16* __restrict__ zn,
                                float* __restrict__ pos,       // [8192]
                                float* __restrict__ partial) { // [256][8][32]
  const int wave = threadIdx.x >> 5;
  const int lane = threadIdx.x & 31;
  const int l16  = lane & 15;
  const int hi   = lane >> 4;

  const int mt = blockIdx.x;
  const int m0 = mt * M_BLOCK;

  // A fragment rows fixed for the whole block (two 16-row tiles).
  const _Float16* arow0 = zn + (size_t)(m0 + l16) * DIM;
  const _Float16* arow1 = arow0 + 16 * DIM;

  float s0[8], s1[8];
#pragma unroll
  for (int r = 0; r < 8; ++r) { s0[r] = 0.f; s1[r] = 0.f; }

  for (int strip = wave; strip < N_STRIPS; strip += WAVES) {
    const int n0 = strip * 64;
    const _Float16* brow = zn + (size_t)(n0 + l16) * DIM;

    v8f c00 = {0.f,0.f,0.f,0.f,0.f,0.f,0.f,0.f};
    v8f c01 = c00, c02 = c00, c03 = c00;
    v8f c10 = c00, c11 = c00, c12 = c00, c13 = c00;

    for (int k0 = 0; k0 < DIM; k0 += 32) {
      const v16h a0 = load_a(arow0, k0, hi);
      const v16h a1 = load_a(arow1, k0, hi);
      const v16h b0 = load_b(brow,            k0, hi);
      const v16h b1 = load_b(brow + 16 * DIM, k0, hi);
      const v16h b2 = load_b(brow + 32 * DIM, k0, hi);
      const v16h b3 = load_b(brow + 48 * DIM, k0, hi);
      c00 = WMMA(a0, b0, c00);  c10 = WMMA(a1, b0, c10);
      c01 = WMMA(a0, b1, c01);  c11 = WMMA(a1, b1, c11);
      c02 = WMMA(a0, b2, c02);  c12 = WMMA(a1, b2, c12);
      c03 = WMMA(a0, b3, c03);  c13 = WMMA(a1, b3, c13);
    }
    // Accumulate exp(logit) per lane for both row tiles.
#pragma unroll
    for (int r = 0; r < 8; ++r) {
      s0[r] += __expf(c00[r] * INV_T) + __expf(c01[r] * INV_T) +
               __expf(c02[r] * INV_T) + __expf(c03[r] * INV_T);
      s1[r] += __expf(c10[r] * INV_T) + __expf(c11[r] * INV_T) +
               __expf(c12[r] * INV_T) + __expf(c13[r] * INV_T);
    }
  }

  // Reduce sum-exp across the 16 column lanes of each half-wave.
#pragma unroll
  for (int r = 0; r < 8; ++r) {
    float v0 = s0[r], v1 = s1[r];
    v0 += __shfl_xor(v0, 1, 32); v1 += __shfl_xor(v1, 1, 32);
    v0 += __shfl_xor(v0, 2, 32); v1 += __shfl_xor(v1, 2, 32);
    v0 += __shfl_xor(v0, 4, 32); v1 += __shfl_xor(v1, 4, 32);
    v0 += __shfl_xor(v0, 8, 32); v1 += __shfl_xor(v1, 8, 32);
    s0[r] = v0; s1[r] = v1;
  }
  // lane 0: rows +0..7 / +16..23; lane 16: rows +8..15 / +24..31.
  if (l16 == 0) {
    float* base = partial + ((size_t)mt * WAVES + wave) * M_BLOCK + hi * 8;
#pragma unroll
    for (int r = 0; r < 8; ++r) { base[r] = s0[r]; base[16 + r] = s1[r]; }
  }

  // Positive logits: recompute the two diagonal tiles (waves 0 and 1).
  if (wave < 2) {
    const int mrow = m0 + wave * 16;                       // tile row base
    const int pos_n0 = ((m0 < B_ROWS) ? m0 : (m0 - B_ROWS)) + wave * 16;
    const _Float16* arow = zn + (size_t)(mrow + l16) * DIM;
    const _Float16* brow = zn + (size_t)(pos_n0 + l16) * DIM;
    v8f c = {0.f,0.f,0.f,0.f,0.f,0.f,0.f,0.f};
#pragma unroll 4
    for (int k0 = 0; k0 < DIM; k0 += 32) {
      c = WMMA(load_a(arow, k0, hi), load_b(brow, k0, hi), c);
    }
#pragma unroll
    for (int r = 0; r < 8; ++r) {
      const float v = c[r] * INV_T;
      if (lane == r)      pos[mrow + r]     = v;  // row mrow+r,   col r
      if (lane == 24 + r) pos[mrow + 8 + r] = v;  // row mrow+8+r, col 8+r
    }
  }
}

// ---------------------------------------------------------------------------
// Kernel 3: loss = mean_i( log(sum_exp_i) - pos_i )
// ---------------------------------------------------------------------------
__global__ void nt_finalize(const float* __restrict__ pos,
                            const float* __restrict__ partial,
                            float* __restrict__ out) {
  __shared__ float red[256];
  const int tid = threadIdx.x;
  float acc = 0.f;
  for (int i = tid; i < NROWS; i += 256) {
    const int mt = i >> 5, r = i & 31;
    const float* base = partial + (size_t)mt * WAVES * M_BLOCK + r;
    float ssum = 0.f;
#pragma unroll
    for (int w = 0; w < WAVES; ++w) ssum += base[w * M_BLOCK];
    acc += logf(ssum) - pos[i];
  }
  red[tid] = acc;
  __syncthreads();
  for (int st = 128; st >= 1; st >>= 1) {
    if (tid < st) red[tid] += red[tid + st];
    __syncthreads();
  }
  if (tid == 0) out[0] = red[0] / (float)NROWS;
}

// ---------------------------------------------------------------------------
extern "C" void kernel_launch(void* const* d_in, const int* in_sizes, int n_in,
                              void* d_out, int out_size, void* d_ws, size_t ws_size,
                              hipStream_t stream) {
  const float* f1 = (const float*)d_in[0];
  const float* f2 = (const float*)d_in[1];
  float* out = (float*)d_out;

  char* ws = (char*)d_ws;
  _Float16* zn    = (_Float16*)ws;                                    // 8 MB
  float*    pos   = (float*)(ws + (size_t)NROWS * DIM * 2);           // 32 KB
  float*    parts = (float*)(ws + (size_t)NROWS * DIM * 2 + NROWS*4); // 256 KB

  nt_normalize   <<<NROWS / WAVES,   256, 0, stream>>>(f1, f2, zn);
  nt_gemm_softmax<<<NROWS / M_BLOCK, 256, 0, stream>>>(zn, pos, parts);
  nt_finalize    <<<1,               256, 0, stream>>>(pos, parts, out);
}
